// GraphSCI_70196945486196
// MI455X (gfx1250) — matrix operand order repaired and checked by reading, not compiled, not run on hardware
//
#include <hip/hip_runtime.h>

// ---------- types for gfx1250 WMMA ----------
typedef __attribute__((ext_vector_type(16))) __bf16       v16bf;
typedef __attribute__((ext_vector_type(8)))  float        v8f;
typedef __attribute__((ext_vector_type(4)))  unsigned int v4u;

struct U32x8 { v4u lo, hi; };

// ---------------------------------------------------------------------------
// GEMM: Out[nrows x 128] = A[nrows x 128] @ W[128 x 128]  (bf16 WMMA, f32 acc)
// MODE 0 (phi):  A is f32.  out0 = acc + bias[n] (f32, ->d_out phi_x);
//                out1 = bf16(scale[row] * out0)   (phi_x_t for next GEMM)
// MODE 1 (gcn):  A is bf16. out0 = scale[row] * acc (f32 h*deg^-1/2)
// Block = 256 threads = 8 waves; each wave computes 16 rows x 128 cols.
// ---------------------------------------------------------------------------
template <int MODE, bool ABF16>
__global__ __launch_bounds__(256) void gemm128_kernel(
    const void* __restrict__ Av, const float* __restrict__ W,
    const float* __restrict__ bias, const float* __restrict__ scale,
    float* __restrict__ out0, __bf16* __restrict__ out1,
    int nrows)
{
  __shared__ __bf16 wl[128 * 136];               // W transposed, n-major, padded

  const int t = threadIdx.x;
  for (int i = t; i < 128 * 128; i += 256) {     // stage W -> LDS as bf16, W^T
    int k = i >> 7, n = i & 127;
    wl[n * 136 + k] = (__bf16)W[i];
  }
  __syncthreads();

  const int lane = t & 31;
  const int wave = t >> 5;
  const long long row0 = (long long)blockIdx.x * 128 + (long long)wave * 16;
  const bool blockFull = ((long long)(blockIdx.x + 1) * 128 <= (long long)nrows);

  // ----- A fragments: 16x32 bf16 per k-step, ISA layout -----
  const int m    = lane & 15;
  const int koff = (lane >> 4) * 8;              // lanes 16-31 start at K=8
  long long row  = row0 + m;
  long long arow = (row < nrows) ? row : (long long)(nrows - 1);

  v16bf a[4];
  if (ABF16) {
    const __bf16* ap = (const __bf16*)Av + arow * 128;
#pragma unroll
    for (int ks = 0; ks < 4; ++ks) {
      U32x8 raw;
      raw.lo = *(const v4u*)(ap + ks * 32 + koff);        // K koff..koff+7
      raw.hi = *(const v4u*)(ap + ks * 32 + koff + 16);   // K koff+16..koff+23
      a[ks] = __builtin_bit_cast(v16bf, raw);
    }
  } else {
    const float* ap = (const float*)Av + arow * 128;
#pragma unroll
    for (int ks = 0; ks < 4; ++ks) {
      const float4 x0 = *(const float4*)(ap + ks * 32 + koff);
      const float4 x1 = *(const float4*)(ap + ks * 32 + koff + 4);
      const float4 x2 = *(const float4*)(ap + ks * 32 + koff + 16);
      const float4 x3 = *(const float4*)(ap + ks * 32 + koff + 20);
      a[ks][0]  = (__bf16)x0.x; a[ks][1]  = (__bf16)x0.y;
      a[ks][2]  = (__bf16)x0.z; a[ks][3]  = (__bf16)x0.w;
      a[ks][4]  = (__bf16)x1.x; a[ks][5]  = (__bf16)x1.y;
      a[ks][6]  = (__bf16)x1.z; a[ks][7]  = (__bf16)x1.w;
      a[ks][8]  = (__bf16)x2.x; a[ks][9]  = (__bf16)x2.y;
      a[ks][10] = (__bf16)x2.z; a[ks][11] = (__bf16)x2.w;
      a[ks][12] = (__bf16)x3.x; a[ks][13] = (__bf16)x3.y;
      a[ks][14] = (__bf16)x3.z; a[ks][15] = (__bf16)x3.w;
    }
  }

  const int n_l = lane & 15;                     // B/D column within tile
  const int kb2 = (lane >> 4) * 8;               // B: lanes 16-31 hold K 8..15, 24..31
  const int vmh = (lane >> 4) * 8;               // D: lanes 16-31 hold M 8..15

  // hoist per-row scale (treatments or deg^-1/2) out of the column-tile loop
  float srow[8];
#pragma unroll
  for (int v = 0; v < 8; ++v) {
    long long r2 = row0 + vmh + v;
    srow[v] = (blockFull || r2 < nrows) ? scale[r2] : 0.0f;
  }

#pragma unroll
  for (int j = 0; j < 8; ++j) {                  // 8 column tiles of 16
    v8f acc;
#pragma unroll
    for (int i = 0; i < 8; ++i) acc[i] = 0.0f;
#pragma unroll
    for (int ks = 0; ks < 4; ++ks) {
      const __bf16* bp = &wl[(j * 16 + n_l) * 136 + ks * 32 + kb2];
      U32x8 raw;
      raw.lo = *(const v4u*)(bp);                // 16B LDS load
      raw.hi = *(const v4u*)(bp + 16);           // 16B LDS load
      v16bf b = __builtin_bit_cast(v16bf, raw);
      acc = __builtin_amdgcn_wmma_f32_16x16x32_bf16(
          false, a[ks], false, b, (short)0, acc, false, false);
    }
    const int n = j * 16 + n_l;
    const float bn = (MODE == 0) ? bias[n] : 0.0f;
    if (blockFull) {                             // uniform fast path
#pragma unroll
      for (int v = 0; v < 8; ++v) {
        long long off = (row0 + vmh + v) * 128 + n;
        if (MODE == 0) {
          float val = acc[v] + bn;
          out0[off] = val;
          out1[off] = (__bf16)(srow[v] * val);
        } else {
          out0[off] = srow[v] * acc[v];
        }
      }
    } else {                                     // boundary block only
#pragma unroll
      for (int v = 0; v < 8; ++v) {
        long long r2 = row0 + vmh + v;
        if (r2 < nrows) {
          long long off = r2 * 128 + n;
          if (MODE == 0) {
            float val = acc[v] + bn;
            out0[off] = val;
            out1[off] = (__bf16)(srow[v] * val);
          } else {
            out0[off] = srow[v] * acc[v];
          }
        }
      }
    }
  }
}

// ---------------------------------------------------------------------------
// Degree kernels
// ---------------------------------------------------------------------------
__global__ void init_deg_kernel(float* __restrict__ deg, int n) {
  int i = blockIdx.x * blockDim.x + threadIdx.x;
  if (i < n) deg[i] = 1.0f;                      // self-loop contribution
}

__global__ void deg_accum_kernel(const long long* __restrict__ dst,
                                 float* __restrict__ deg, long long E) {
  long long e = (long long)blockIdx.x * blockDim.x + threadIdx.x;
  if (e < E) atomicAdd(&deg[dst[e]], 1.0f);
}

__global__ void deg_rsqrt_kernel(float* __restrict__ deg, int n) {
  int i = blockIdx.x * blockDim.x + threadIdx.x;
  if (i < n) deg[i] = rsqrtf(deg[i]);            // deg >= 1 always
}

// ---------------------------------------------------------------------------
// Scatter: agg[dst] += h[src]   (one wave per edge, float4 per lane)
// ---------------------------------------------------------------------------
__global__ __launch_bounds__(256) void scatter_kernel(
    const long long* __restrict__ src, const long long* __restrict__ dst,
    const float* __restrict__ h, float* __restrict__ agg, long long E)
{
  long long tid = (long long)blockIdx.x * blockDim.x + threadIdx.x;
  long long e = tid >> 5;
  if (e >= E) return;
  int c = (int)(tid & 31) * 4;
  long long s = src[e], d = dst[e];
  float4 v = *(const float4*)(h + s * 128 + c);
  float* o = agg + d * 128 + c;
  atomicAdd(o + 0, v.x);
  atomicAdd(o + 1, v.y);
  atomicAdd(o + 2, v.z);
  atomicAdd(o + 3, v.w);
}

// ---------------------------------------------------------------------------
// Finalize conv1: rep1 = relu( dis[row]*(agg + hs) + b[n] )  -> bf16 for GEMM2
// ---------------------------------------------------------------------------
__global__ __launch_bounds__(256) void finalize_kernel(
    const float* __restrict__ agg, const float* __restrict__ hs,
    const float* __restrict__ dis, const float* __restrict__ b,
    __bf16* __restrict__ out, long long total)
{
  long long i = (long long)blockIdx.x * blockDim.x + threadIdx.x;
  if (i >= total) return;
  long long row = i >> 7;
  int n = (int)(i & 127);
  float v = dis[row] * (agg[i] + hs[i]) + b[n];
  out[i] = (__bf16)fmaxf(v, 0.0f);
}

// ---------------------------------------------------------------------------
// Fused conv2-finalize + heads: one wave per row.
// rep = relu(dis*(agg2+h2s)+b2);  y0 = rep.w01[128:]+b01;
// y1 = phi_x.w11[:128] + rep.w11[128:] + b11
// ---------------------------------------------------------------------------
__global__ __launch_bounds__(256) void head_kernel(
    const float* __restrict__ agg, const float* __restrict__ hs,
    const float* __restrict__ dis, const float* __restrict__ b2,
    const float* __restrict__ phi_x,
    const float* __restrict__ w01, const float* __restrict__ b01,
    const float* __restrict__ w11, const float* __restrict__ b11,
    float* __restrict__ y1, float* __restrict__ y0, int nrows)
{
  int wave = threadIdx.x >> 5, lane = threadIdx.x & 31;
  long long row = (long long)blockIdx.x * 8 + wave;
  if (row >= nrows) return;
  int c = lane * 4;
  long long off = row * 128 + c;
  float4 ag = *(const float4*)(agg + off);
  float4 h  = *(const float4*)(hs + off);
  float4 px = *(const float4*)(phi_x + off);
  float4 bb = *(const float4*)(b2 + c);
  float dr = dis[row];
  float r0 = fmaxf(dr * (ag.x + h.x) + bb.x, 0.0f);
  float r1 = fmaxf(dr * (ag.y + h.y) + bb.y, 0.0f);
  float r2 = fmaxf(dr * (ag.z + h.z) + bb.z, 0.0f);
  float r3 = fmaxf(dr * (ag.w + h.w) + bb.w, 0.0f);
  float4 w01h = *(const float4*)(w01 + 128 + c);
  float4 w11l = *(const float4*)(w11 + c);
  float4 w11h = *(const float4*)(w11 + 128 + c);
  float a0 = r0 * w01h.x + r1 * w01h.y + r2 * w01h.z + r3 * w01h.w;
  float a1 = px.x * w11l.x + px.y * w11l.y + px.z * w11l.z + px.w * w11l.w
           + r0 * w11h.x + r1 * w11h.y + r2 * w11h.z + r3 * w11h.w;
#pragma unroll
  for (int s2 = 16; s2 > 0; s2 >>= 1) {
    a0 += __shfl_xor(a0, s2, 32);
    a1 += __shfl_xor(a1, s2, 32);
  }
  if (lane == 0) {
    y0[row] = a0 + b01[0];
    y1[row] = a1 + b11[0];
  }
}

// ---------------------------------------------------------------------------
extern "C" void kernel_launch(void* const* d_in, const int* in_sizes, int n_in,
                              void* d_out, int out_size, void* d_ws, size_t ws_size,
                              hipStream_t stream) {
  const float*     features   = (const float*)d_in[0];
  const float*     treatments = (const float*)d_in[1];
  const long long* edges      = (const long long*)d_in[2];
  const float*     phi_w = (const float*)d_in[3];
  const float*     phi_b = (const float*)d_in[4];
  const float*     g1w   = (const float*)d_in[5];
  const float*     g1b   = (const float*)d_in[6];
  const float*     g2w   = (const float*)d_in[7];
  const float*     g2b   = (const float*)d_in[8];
  const float*     w01   = (const float*)d_in[9];
  const float*     b01   = (const float*)d_in[10];
  const float*     w11   = (const float*)d_in[11];
  const float*     b11   = (const float*)d_in[12];

  const int n = in_sizes[0] / 128;
  const long long E = (long long)in_sizes[2] / 2;
  const long long* src = edges;
  const long long* dst = edges + E;

  float* y1    = (float*)d_out;
  float* y0    = y1 + n;
  float* phi_x = y0 + n;                          // N x 128 region of d_out

  float*  dis   = (float*)d_ws;                   // N       : deg -> deg^{-1/2}
  __bf16* buf0b = (__bf16*)(dis + n);             // N x 128 bf16: phi_x_t / rep1
  float*  buf1  = (float*)(dis + n) + (size_t)n * 128;  // N x 128 f32: h1s/h2s
  float*  buf2  = buf1 + (size_t)n * 128;         // N x 128 f32: scatter acc

  const int B = 256;
  const unsigned nb   = (unsigned)((n + B - 1) / B);
  const unsigned eb   = (unsigned)((E + B - 1) / B);
  const unsigned gb   = (unsigned)((n + 127) / 128);          // GEMM blocks
  const unsigned sb   = (unsigned)((E * 32 + B - 1) / B);     // scatter blocks
  const long long tot = (long long)n * 128;
  const unsigned fb   = (unsigned)((tot + B - 1) / B);

  // --- degrees: deg = 1 (self loop) + scatter ones; dis = rsqrt(deg) ---
  init_deg_kernel<<<nb, B, 0, stream>>>(dis, n);
  deg_accum_kernel<<<eb, B, 0, stream>>>(dst, dis, E);
  deg_rsqrt_kernel<<<nb, B, 0, stream>>>(dis, n);

  // --- phi: phi_x = X @ phi_w + b (-> d_out f32), phi_x_t -> bf16 buf0 ---
  gemm128_kernel<0, false><<<gb, B, 0, stream>>>(
      features, phi_w, phi_b, treatments, phi_x, buf0b, n);

  // --- conv1: h1s = dis * (phi_x_t @ g1w); scatter; relu-finalize -> bf16 ---
  hipMemsetAsync(buf2, 0, (size_t)n * 128 * sizeof(float), stream);
  gemm128_kernel<1, true><<<gb, B, 0, stream>>>(
      buf0b, g1w, nullptr, dis, buf1, nullptr, n);
  scatter_kernel<<<sb, B, 0, stream>>>(src, dst, buf1, buf2, E);
  finalize_kernel<<<fb, B, 0, stream>>>(buf2, buf1, dis, g1b, buf0b, tot);

  // --- conv2: h2s = dis * (rep1 @ g2w); scatter; fused finalize + heads ---
  gemm128_kernel<1, true><<<gb, B, 0, stream>>>(
      buf0b, g2w, nullptr, dis, buf1, nullptr, n);
  hipMemsetAsync(buf2, 0, (size_t)n * 128 * sizeof(float), stream);
  scatter_kernel<<<sb, B, 0, stream>>>(src, dst, buf1, buf2, E);
  head_kernel<<<(unsigned)((n + 7) / 8), B, 0, stream>>>(
      buf2, buf1, dis, g2b, phi_x, w01, b01, w11, b11, y1, y0, n);
}
